// CausalSelfAttention_3075196584190
// MI455X (gfx1250) — matrix operand order repaired and checked by reading, not compiled, not run on hardware
//
#include <hip/hip_runtime.h>
#include <hip/hip_bf16.h>

#define BB   4
#define TT   2048
#define DD   1024
#define HH   16
#define DHH  64
#define MROWS (BB*TT)     // 8192
#define N3D   (3*DD)      // 3072
#define KC    32          // K-chunk per WMMA pass
#define PITCH 40          // LDS row pitch in elements (32 + 8 pad -> conflict-free b128)
#define ASTRIDE (256 * PITCH)   // elements per A buffer
#define BSTRIDE (128 * PITCH)   // elements per B buffer

typedef __bf16 bf16_t;
typedef __attribute__((ext_vector_type(16))) __bf16 v16bf;
typedef __attribute__((ext_vector_type(8)))  __bf16 v8bf;
typedef __attribute__((ext_vector_type(8)))  float  v8f;

static __device__ __forceinline__ v8f wmma_bf16(v16bf a, v16bf b, v8f c) {
  // D = A(16x32 bf16) * B(32x16 bf16) + C(16x16 f32)
  return __builtin_amdgcn_wmma_f32_16x16x32_bf16(false, a, false, b, (short)0, c,
                                                 false, false);
}

// Build a 16-element bf16 fragment from two contiguous 8-element (16B) runs.
static __device__ __forceinline__ v16bf ldfrag(const bf16_t* p0, const bf16_t* p1) {
  union { v16bf v; v8bf h[2]; } u;
  u.h[0] = *(const v8bf*)p0;
  u.h[1] = *(const v8bf*)p1;
  return u.v;
}

// Async copy 16B from global to LDS (per lane). Tracked by ASYNCcnt.
static __device__ __forceinline__ void async_b128(uint32_t lds_off, const bf16_t* g) {
  asm volatile("global_load_async_to_lds_b128 %0, %1, off"
               :: "v"(lds_off), "v"((unsigned long long)(uintptr_t)g)
               : "memory");
}
static __device__ __forceinline__ void wait_async_le6() {
  asm volatile("s_wait_asynccnt 6" ::: "memory");
}
static __device__ __forceinline__ void wait_async_0() {
  asm volatile("s_wait_asynccnt 0" ::: "memory");
}

// ---------------------------------------------------------------- convert
__global__ void f32_to_bf16_kernel(const float* __restrict__ src,
                                   bf16_t* __restrict__ dst, int n) {
  int i = (blockIdx.x * 256 + threadIdx.x) * 4;
  if (i + 3 < n) {
    float4 f = *(const float4*)(src + i);
    dst[i + 0] = (bf16_t)f.x;
    dst[i + 1] = (bf16_t)f.y;
    dst[i + 2] = (bf16_t)f.z;
    dst[i + 3] = (bf16_t)f.w;
  }
}

// ---------------------------------------------------------------- QKV GEMM
// C[8192,3072] = Xb[8192,1024] * Wattn^T   (Wattn row-major [3072,1024])
// 8-wave workgroup on a 256x128 macro tile; LDS double-buffered K-chunks
// staged with global_load_async_to_lds_b128; each wave computes 64x64.
__global__ __launch_bounds__(256) void qkv_gemm_kernel(
    const bf16_t* __restrict__ xb, const bf16_t* __restrict__ wab,
    bf16_t* __restrict__ qb, bf16_t* __restrict__ kbuf, bf16_t* __restrict__ vT) {
  __shared__ __align__(16) bf16_t As[2 * ASTRIDE];   // 2 x 20480 B
  __shared__ __align__(16) bf16_t Bs[2 * BSTRIDE];   // 2 x 10240 B

  const int tid  = threadIdx.x;
  const int lane = tid & 31;
  const int wid  = tid >> 5;        // 0..7
  const int l15  = lane & 15;
  const bool hi  = lane >= 16;
  const int tx = blockIdx.x % (N3D / 128);
  const int ty = blockIdx.x / (N3D / 128);
  const int row0 = ty * 256;
  const int col0 = tx * 128;

  // -- staging source/dest per thread
  const bf16_t* gA = xb  + (size_t)(row0 + tid) * DD;         // one A row / thread
  const bf16_t* gB = wab + (size_t)(col0 + (tid >> 1)) * DD;  // one B row / 2 threads
  const int bc0 = (tid & 1) * 2;                              // B chunk pair 0-1 / 2-3
  const uint32_t ldsA0 = (uint32_t)(uintptr_t)&As[0] + tid * (PITCH * 2);
  const uint32_t ldsB0 = (uint32_t)(uintptr_t)&Bs[0] + (tid >> 1) * (PITCH * 2)
                         + bc0 * 16;

  // -- compute-side fragment offsets (2x4 wave grid over 256x128)
  const int wy = wid >> 1;          // 0..3 -> row block of 64
  const int wx = wid & 1;           // 0..1 -> col block of 64
  const int aoff = hi ? 8 : 0;
  const int boff = hi ? 8 : 0;

  v8f acc[4][4];
#pragma unroll
  for (int i = 0; i < 4; i++)
#pragma unroll
    for (int j = 0; j < 4; j++) acc[i][j] = (v8f)0.0f;

  // issue async copies for chunk starting at k0 into buffer `s`
  auto issue = [&](int k0, int s) {
    const uint32_t la = ldsA0 + s * (ASTRIDE * 2);
    const uint32_t lb = ldsB0 + s * (BSTRIDE * 2);
#pragma unroll
    for (int c2 = 0; c2 < 4; c2++)
      async_b128(la + c2 * 16, gA + k0 + c2 * 8);
#pragma unroll
    for (int c2 = 0; c2 < 2; c2++)
      async_b128(lb + c2 * 16, gB + k0 + (bc0 + c2) * 8);
  };  // 6 async instructions per wave

  // compute one K-chunk from buffer `buf` (direct GEPs -> ds_load_b128)
  auto compute = [&](int buf) {
    const int ab = buf * ASTRIDE;
    const int bb = buf * BSTRIDE;
    v16bf a[4], b[4];
#pragma unroll
    for (int i = 0; i < 4; i++) {
      const int o = ab + (wy*64 + 16*i + l15) * PITCH + aoff;
      a[i] = ldfrag(&As[o], &As[o + 16]);
    }
#pragma unroll
    for (int j = 0; j < 4; j++) {
      const int o = bb + (wx*64 + 16*j + l15) * PITCH + boff;
      b[j] = ldfrag(&Bs[o], &Bs[o + 16]);
    }
#pragma unroll
    for (int i = 0; i < 4; i++)
#pragma unroll
      for (int j = 0; j < 4; j++)
        acc[i][j] = wmma_bf16(a[i], b[j], acc[i][j]);
  };

  issue(0, 0);
  int buf = 0;
  for (int c = 0; c < DD - KC; c += KC) {   // steady state: branch-free body
    issue(c + KC, buf ^ 1);
    wait_async_le6();
    __syncthreads();
    compute(buf);
    __syncthreads();
    buf ^= 1;
  }
  wait_async_0();                            // peeled final chunk
  __syncthreads();
  compute(buf);

  // Epilogue: scatter into q/k/vT (bf16). C layout: VGPR r -> M=r / M=8+r.
  const int rbase = hi ? 8 : 0;
#pragma unroll
  for (int j = 0; j < 4; j++) {
    const int e   = col0 + wx*64 + 16*j + l15;   // [0,3072)
    const int s   = e >> 10;                     // 0=q 1=k 2=v
    const int rem = e & 1023;
    const int h   = rem >> 6;
    const int dh  = rem & 63;
#pragma unroll
    for (int i = 0; i < 4; i++) {
#pragma unroll
      for (int r = 0; r < 8; r++) {
        const int grow = row0 + wy*64 + 16*i + r + rbase;
        const int b    = grow >> 11;
        const int t    = grow & (TT - 1);
        const int bh   = b * HH + h;
        const bf16_t val = (bf16_t)acc[i][j][r];
        if (s == 0)      qb  [((size_t)bh*TT + t)*DHH + dh] = val;
        else if (s == 1) kbuf[((size_t)bh*TT + t)*DHH + dh] = val;
        else             vT  [((size_t)bh*DHH + dh)*TT + t] = val;
      }
    }
  }
}

// ---------------------------------------------------------------- attention
// One wave per (b,h,16-row q tile). Flash-style online softmax, key blocks
// of 32, S and P*V via WMMA, P re-laid-out through LDS.
__global__ __launch_bounds__(32) void attn_kernel(
    const bf16_t* __restrict__ qb, const bf16_t* __restrict__ kbuf,
    const bf16_t* __restrict__ vT, bf16_t* __restrict__ yb) {
  const int lane = threadIdx.x;
  const int l15  = lane & 15;
  const bool hi  = lane >= 16;
  const int qt = blockIdx.x & (TT/16 - 1);
  const int bh = blockIdx.x / (TT/16);
  const int q0 = qt * 16;
  const int rbase = hi ? 8 : 0;

  __shared__ __align__(16) bf16_t Pl[16 * 32];

  const bf16_t* qrow = qb + ((size_t)bh*TT + q0 + l15) * DHH;
  v16bf qf[2];
#pragma unroll
  for (int c = 0; c < 2; c++) {
    const bf16_t* p = qrow + c*32 + (hi ? 8 : 0);
    qf[c] = ldfrag(p, p + 16);
  }

  v8f O[4];
#pragma unroll
  for (int j = 0; j < 4; j++) O[j] = (v8f)0.0f;
  float m[8], lsum[8];
#pragma unroll
  for (int r = 0; r < 8; r++) { m[r] = -1.0e30f; lsum[r] = 0.0f; }

  for (int k0 = 0; k0 < q0 + 16; k0 += 32) {
    v8f S[2];
    S[0] = (v8f)0.0f; S[1] = (v8f)0.0f;
#pragma unroll
    for (int tN = 0; tN < 2; tN++) {
#pragma unroll
      for (int c = 0; c < 2; c++) {
        const bf16_t* kp = kbuf + ((size_t)bh*TT + k0 + tN*16 + l15) * DHH
                                + c*32 + (hi ? 8 : 0);
        v16bf kf = ldfrag(kp, kp + 16);
        S[tN] = wmma_bf16(qf[c], kf, S[tN]);
      }
    }
    const float sc = 0.125f;   // 1/sqrt(64)
    if (k0 + 31 > q0) {
#pragma unroll
      for (int tN = 0; tN < 2; tN++)
#pragma unroll
        for (int r = 0; r < 8; r++) {
          const int key = k0 + tN*16 + l15;
          const int row = q0 + r + rbase;
          S[tN][r] = (key <= row) ? S[tN][r] * sc : -1.0e30f;
        }
    } else {
#pragma unroll
      for (int tN = 0; tN < 2; tN++)
#pragma unroll
        for (int r = 0; r < 8; r++) S[tN][r] *= sc;
    }

    float alpha[8];
#pragma unroll
    for (int r = 0; r < 8; r++) {
      float v = fmaxf(S[0][r], S[1][r]);
#pragma unroll
      for (int off = 1; off < 16; off <<= 1)
        v = fmaxf(v, __shfl_xor(v, off, 16));
      const float mn = fmaxf(m[r], v);
      alpha[r] = __expf(m[r] - mn);
      m[r] = mn;
    }
#pragma unroll
    for (int r = 0; r < 8; r++) {
      const float p0 = __expf(S[0][r] - m[r]);
      const float p1 = __expf(S[1][r] - m[r]);
      Pl[(r + rbase) * 32 + l15]      = (bf16_t)p0;
      Pl[(r + rbase) * 32 + 16 + l15] = (bf16_t)p1;
      float s = p0 + p1;
#pragma unroll
      for (int off = 1; off < 16; off <<= 1)
        s += __shfl_xor(s, off, 16);
      lsum[r] = lsum[r] * alpha[r] + s;
#pragma unroll
      for (int j = 0; j < 4; j++) O[j][r] *= alpha[r];
    }
    __syncthreads();

    const int po = l15 * 32 + (hi ? 8 : 0);
    v16bf pf = ldfrag(&Pl[po], &Pl[po + 16]);
#pragma unroll
    for (int j = 0; j < 4; j++) {
      const bf16_t* vp = vT + ((size_t)bh*DHH + 16*j + l15) * TT
                            + k0 + (hi ? 8 : 0);
      v16bf vf = ldfrag(vp, vp + 16);
      O[j] = wmma_bf16(pf, vf, O[j]);
    }
    __syncthreads();
  }

  const int b = bh / HH;
  const int h = bh % HH;
#pragma unroll
  for (int r = 0; r < 8; r++) {
    const float inv = 1.0f / lsum[r];
    const int row = q0 + r + rbase;
#pragma unroll
    for (int j = 0; j < 4; j++) {
      const int dh = 16*j + l15;
      yb[(((size_t)b*TT + row)*HH + h)*DHH + dh] = (bf16_t)(O[j][r] * inv);
    }
  }
}

// ---------------------------------------------------------------- projection
// out[8192,1024] (f32) = Yb[8192,1024] * Wproj^T. Same LDS-staged scheme.
__global__ __launch_bounds__(256) void proj_gemm_kernel(
    const bf16_t* __restrict__ yb, const bf16_t* __restrict__ wpb,
    float* __restrict__ out) {
  __shared__ __align__(16) bf16_t As[2 * ASTRIDE];
  __shared__ __align__(16) bf16_t Bs[2 * BSTRIDE];

  const int tid  = threadIdx.x;
  const int lane = tid & 31;
  const int wid  = tid >> 5;
  const int l15  = lane & 15;
  const bool hi  = lane >= 16;
  const int tx = blockIdx.x % (DD / 128);
  const int ty = blockIdx.x / (DD / 128);
  const int row0 = ty * 256;
  const int col0 = tx * 128;

  const bf16_t* gA = yb  + (size_t)(row0 + tid) * DD;
  const bf16_t* gB = wpb + (size_t)(col0 + (tid >> 1)) * DD;
  const int bc0 = (tid & 1) * 2;
  const uint32_t ldsA0 = (uint32_t)(uintptr_t)&As[0] + tid * (PITCH * 2);
  const uint32_t ldsB0 = (uint32_t)(uintptr_t)&Bs[0] + (tid >> 1) * (PITCH * 2)
                         + bc0 * 16;

  const int wy = wid >> 1;
  const int wx = wid & 1;
  const int aoff = hi ? 8 : 0;
  const int boff = hi ? 8 : 0;

  v8f acc[4][4];
#pragma unroll
  for (int i = 0; i < 4; i++)
#pragma unroll
    for (int j = 0; j < 4; j++) acc[i][j] = (v8f)0.0f;

  auto issue = [&](int k0, int s) {
    const uint32_t la = ldsA0 + s * (ASTRIDE * 2);
    const uint32_t lb = ldsB0 + s * (BSTRIDE * 2);
#pragma unroll
    for (int c2 = 0; c2 < 4; c2++)
      async_b128(la + c2 * 16, gA + k0 + c2 * 8);
#pragma unroll
    for (int c2 = 0; c2 < 2; c2++)
      async_b128(lb + c2 * 16, gB + k0 + (bc0 + c2) * 8);
  };

  auto compute = [&](int buf) {
    const int ab = buf * ASTRIDE;
    const int bb = buf * BSTRIDE;
    v16bf a[4], b[4];
#pragma unroll
    for (int i = 0; i < 4; i++) {
      const int o = ab + (wy*64 + 16*i + l15) * PITCH + aoff;
      a[i] = ldfrag(&As[o], &As[o + 16]);
    }
#pragma unroll
    for (int j = 0; j < 4; j++) {
      const int o = bb + (wx*64 + 16*j + l15) * PITCH + boff;
      b[j] = ldfrag(&Bs[o], &Bs[o + 16]);
    }
#pragma unroll
    for (int i = 0; i < 4; i++)
#pragma unroll
      for (int j = 0; j < 4; j++)
        acc[i][j] = wmma_bf16(a[i], b[j], acc[i][j]);
  };

  issue(0, 0);
  int buf = 0;
  for (int c = 0; c < DD - KC; c += KC) {
    issue(c + KC, buf ^ 1);
    wait_async_le6();
    __syncthreads();
    compute(buf);
    __syncthreads();
    buf ^= 1;
  }
  wait_async_0();
  __syncthreads();
  compute(buf);

  const int rbase = hi ? 8 : 0;
#pragma unroll
  for (int j = 0; j < 4; j++) {
    const int e = col0 + wx*64 + 16*j + l15;
#pragma unroll
    for (int i = 0; i < 4; i++) {
#pragma unroll
      for (int r = 0; r < 8; r++) {
        const int grow = row0 + wy*64 + 16*i + r + rbase;
        out[(size_t)grow * DD + e] = acc[i][j][r];
      }
    }
  }
}

// ---------------------------------------------------------------- launcher
extern "C" void kernel_launch(void* const* d_in, const int* in_sizes, int n_in,
                              void* d_out, int out_size, void* d_ws, size_t ws_size,
                              hipStream_t stream) {
  const float* x      = (const float*)d_in[0];
  const float* w_attn = (const float*)d_in[1];
  const float* w_proj = (const float*)d_in[2];
  float* out = (float*)d_out;

  char* ws = (char*)d_ws;
  size_t off = 0;
  bf16_t* xb   = (bf16_t*)(ws + off); off += (size_t)MROWS * DD * 2;
  bf16_t* wab  = (bf16_t*)(ws + off); off += (size_t)N3D   * DD * 2;
  bf16_t* wpb  = (bf16_t*)(ws + off); off += (size_t)DD    * DD * 2;
  bf16_t* qb   = (bf16_t*)(ws + off); off += (size_t)BB*HH*TT*DHH * 2;
  bf16_t* kbuf = (bf16_t*)(ws + off); off += (size_t)BB*HH*TT*DHH * 2;
  bf16_t* vT   = (bf16_t*)(ws + off); off += (size_t)BB*HH*TT*DHH * 2;
  bf16_t* yb   = (bf16_t*)(ws + off); off += (size_t)MROWS * DD * 2;

  f32_to_bf16_kernel<<<(MROWS*DD)/1024, 256, 0, stream>>>(x, xb, MROWS*DD);
  f32_to_bf16_kernel<<<(N3D*DD)/1024,   256, 0, stream>>>(w_attn, wab, N3D*DD);
  f32_to_bf16_kernel<<<(DD*DD)/1024,    256, 0, stream>>>(w_proj, wpb, DD*DD);

  qkv_gemm_kernel<<<(MROWS/256)*(N3D/128), 256, 0, stream>>>(xb, wab, qb, kbuf, vT);
  attn_kernel<<<BB*HH*(TT/16), 32, 0, stream>>>(qb, kbuf, vT, yb);
  proj_gemm_kernel<<<(MROWS/256)*(DD/128), 256, 0, stream>>>(yb, wpb, out);
}
